// IJEPA_fMRI_base_36258113913327
// MI455X (gfx1250) — matrix operand order repaired and checked by reading, not compiled, not run on hardware
//
#include <hip/hip_runtime.h>
#include <cstddef>

// ---------------------------------------------------------------------------
// Types for CDNA5 WMMA (wave32): v_wmma_f32_16x16x32_bf16
// ---------------------------------------------------------------------------
typedef __attribute__((ext_vector_type(16))) __bf16 bf16x16;
typedef __attribute__((ext_vector_type(8)))  float  floatx8;

__device__ inline floatx8 wmma_bf16(bf16x16 a, bf16x16 b, floatx8 c) {
  // (neg_a, A, neg_b, B, c_mod, C, reuse_a, reuse_b)
  return __builtin_amdgcn_wmma_f32_16x16x32_bf16(false, a, false, b, (short)0, c,
                                                 false, false);
}

__device__ inline __bf16 to_bf16(float f) { return (__bf16)f; }

__device__ inline unsigned short bf16_bits(float f) {
  union { __bf16 b; unsigned short u; } t;
  t.b = (__bf16)f;
  return t.u;
}
// pack two fp32 -> one dword holding {lo:bf16, hi:bf16}
__device__ inline unsigned int pack2_bf16(float lo, float hi) {
  return (unsigned int)bf16_bits(lo) | ((unsigned int)bf16_bits(hi) << 16);
}

// Fragment unions: force wide DS loads.
union AFrag { bf16x16 v; uint2 d[4]; };   // two contiguous 16B runs
union BFragW { bf16x16 v; uint4 q[2]; };  // two runs of 4 pair-dwords (p=4h.., 8+4h..)
union BFragR { bf16x16 v; uint2 d[4]; };  // two 16B runs of a row-major row

__device__ inline float gelu_tanh(float x) {
  const float k0 = 0.7978845608028654f;   // sqrt(2/pi)
  const float k1 = 0.044715f;
  float x3 = x * x * x;
  return 0.5f * x * (1.0f + tanhf(k0 * (x + k1 * x3)));
}

__device__ inline float spike(float v) { return v > 1.0f ? 1.0f : 0.0f; }

// ---------------------------------------------------------------------------
// Generic fused GEMM:  C[M,N] = epilogue( A[M,K] @ W[K,N] )
//   epilogue: (+bias[col]) (+pos[(row%posR)*N+col]) (gelu) (+resid) (spike)
// 64x64 tile / block, 4 waves; A staged row-major bf16 (stride 36),
// W staged pair-interleaved dwords (stride 36 dwords per column).
// ---------------------------------------------------------------------------
__global__ __launch_bounds__(128) void gemm_kernel(
    const float* __restrict__ A, const float* __restrict__ W,
    float* __restrict__ C, int M, int N, int K,
    const float* __restrict__ bias,
    const float* __restrict__ pos, int posR,
    const float* __restrict__ resid,
    int doGelu, int doSpike)
{
  __shared__ __align__(16) __bf16      As[64 * 36];     // [row][k] stride 36
  __shared__ __align__(16) unsigned int Bs2[64 * 36];   // [n][kpair] stride 36

  const int tid  = threadIdx.x;
  const int wave = tid >> 5;
  const int lane = tid & 31;
  const int l    = lane & 15;
  const int hi   = lane >> 4;
  const int m0   = blockIdx.x * 64;
  const int n0   = blockIdx.y * 64;

  floatx8 acc[4];
#pragma unroll
  for (int j = 0; j < 4; ++j)
#pragma unroll
    for (int e = 0; e < 8; ++e) acc[j][e] = 0.0f;

  const int arow = tid >> 1;           // 0..63
  const int acol = (tid & 1) * 16;     // 0 / 16
  const int bpr  = tid >> 3;           // 0..15  (k-pair row)
  const int bc8  = (tid & 7) * 8;      // 0..56  (8 columns)

  for (int k0 = 0; k0 < K; k0 += 32) {
    // ---- stage A tile: fp32 -> bf16, packed dword stores ----
    {
      const int gm = m0 + arow;
      if (gm < M && k0 + 64 < K)
        __builtin_prefetch(A + (size_t)gm * K + k0 + 32 + acol, 0, 1);
#pragma unroll
      for (int c = 0; c < 16; c += 4) {
        const int gk = k0 + acol + c;
        float t[4] = {0.f, 0.f, 0.f, 0.f};
        if (gm < M) {
          if (gk + 3 < K) {
            const float4 f = *(const float4*)(A + (size_t)gm * K + gk);
            t[0] = f.x; t[1] = f.y; t[2] = f.z; t[3] = f.w;
          } else {
#pragma unroll
            for (int i = 0; i < 4; ++i)
              if (gk + i < K) t[i] = A[(size_t)gm * K + gk + i];
          }
        }
        *(unsigned int*)&As[arow * 36 + acol + c]     = pack2_bf16(t[0], t[1]);
        *(unsigned int*)&As[arow * 36 + acol + c + 2] = pack2_bf16(t[2], t[3]);
      }
    }
    // ---- stage W tile pair-interleaved: Bs2[n][p] = {W[k0+2p][n], W[k0+2p+1][n]}
    {
      const int gk0 = k0 + 2 * bpr;
      float w0[8], w1[8];
#pragma unroll
      for (int c = 0; c < 8; c += 4) {
        const int gn = n0 + bc8 + c;
        float a0[4] = {0.f, 0.f, 0.f, 0.f};
        float a1[4] = {0.f, 0.f, 0.f, 0.f};
        if (gk0 < K) {
          if (gn + 3 < N) {
            const float4 f = *(const float4*)(W + (size_t)gk0 * N + gn);
            a0[0] = f.x; a0[1] = f.y; a0[2] = f.z; a0[3] = f.w;
          } else {
#pragma unroll
            for (int i = 0; i < 4; ++i)
              if (gn + i < N) a0[i] = W[(size_t)gk0 * N + gn + i];
          }
        }
        if (gk0 + 1 < K) {
          if (gn + 3 < N) {
            const float4 f = *(const float4*)(W + (size_t)(gk0 + 1) * N + gn);
            a1[0] = f.x; a1[1] = f.y; a1[2] = f.z; a1[3] = f.w;
          } else {
#pragma unroll
            for (int i = 0; i < 4; ++i)
              if (gn + i < N) a1[i] = W[(size_t)(gk0 + 1) * N + gn + i];
          }
        }
#pragma unroll
        for (int i = 0; i < 4; ++i) { w0[c + i] = a0[i]; w1[c + i] = a1[i]; }
      }
#pragma unroll
      for (int c = 0; c < 8; ++c)
        Bs2[(bc8 + c) * 36 + bpr] = pack2_bf16(w0[c], w1[c]);
    }
    __syncthreads();

    // ---- A fragment: 4x ds_load_b64 ----
    AFrag af;
    {
      const __bf16* ap = As + (16 * wave + l) * 36 + hi * 8;
      af.d[0] = *(const uint2*)(ap);
      af.d[1] = *(const uint2*)(ap + 4);
      af.d[2] = *(const uint2*)(ap + 16);
      af.d[3] = *(const uint2*)(ap + 20);
    }
    // ---- 4 N-subtiles, B fragment: 2x ds_load_b128 each ----
#pragma unroll
    for (int j = 0; j < 4; ++j) {
      BFragW bfr;
      const unsigned int* bp = Bs2 + (16 * j + l) * 36 + hi * 4;
      bfr.q[0] = *(const uint4*)(bp);
      bfr.q[1] = *(const uint4*)(bp + 8);
      acc[j] = wmma_bf16(af.v, bfr.v, acc[j]);
    }
    __syncthreads();
  }

  // ---- fused epilogue + store ----
#pragma unroll
  for (int j = 0; j < 4; ++j) {
#pragma unroll
    for (int v = 0; v < 8; ++v) {
      const int row = m0 + 16 * wave + 8 * hi + v;
      const int col = n0 + 16 * j + l;
      if (row < M && col < N) {
        float val = acc[j][v];
        if (bias)  val += bias[col];
        if (pos)   val += pos[(size_t)(row % posR) * N + col];
        if (doGelu)  val = gelu_tanh(val);
        if (resid) val += resid[(size_t)row * N + col];
        if (doSpike) val = spike(val);
        C[(size_t)row * N + col] = val;
      }
    }
  }
}

// ---------------------------------------------------------------------------
// Fused multi-head attention, one block per (sequence, head).
// Tokens 90 (padded 96), hd=32, H=8. qkv row: [seq*90+tok, 768].
// ---------------------------------------------------------------------------
__global__ __launch_bounds__(192) void attention_kernel(
    const float* __restrict__ qkv, float* __restrict__ out, int causal)
{
  const int s = blockIdx.x;
  const int h = blockIdx.y;

  __shared__ __align__(16) __bf16       Qs[96 * 36];   // [tok][d]   stride 36
  __shared__ __align__(16) __bf16       Ks[96 * 36];   // [tok][d]   stride 36
  __shared__ __align__(16) unsigned int Vs2[32 * 52];  // [d][tokpair] stride 52
  __shared__ __align__(16) __bf16       Ps[96 * 104];  // [row][col] stride 104

  const int tid  = threadIdx.x;
  const int wave = tid >> 5;           // 0..5
  const int lane = tid & 31;
  const int l    = lane & 15;
  const int hi   = lane >> 4;

  // ---- stage Q, K row-major (packed dwords); V pair-interleaved ----
  {
    const int r  = tid >> 1;           // 0..95
    const int c0 = (tid & 1) * 16;
    const float* base = qkv + ((size_t)s * 90) * 768 + h * 32;
#pragma unroll
    for (int mat = 0; mat < 2; ++mat) {           // 0:Q 1:K
      __bf16* dst = (mat == 0) ? Qs : Ks;
      const float* src = base + mat * 256;
#pragma unroll
      for (int c = 0; c < 16; c += 2) {
        const float e0 = (r < 90) ? src[(size_t)r * 768 + c0 + c]     : 0.0f;
        const float e1 = (r < 90) ? src[(size_t)r * 768 + c0 + c + 1] : 0.0f;
        *(unsigned int*)&dst[r * 36 + c0 + c] = pack2_bf16(e0, e1);
      }
    }
    // V: pair token rows 2p, 2p+1; dims c8..c8+7
    const int p  = tid >> 2;           // 0..47
    const int c8 = (tid & 3) * 8;
    const float* vsrc = base + 512;
    float v0[8], v1[8];
#pragma unroll
    for (int c = 0; c < 8; ++c) {
      const int t0 = 2 * p, t1 = 2 * p + 1;
      v0[c] = (t0 < 90) ? vsrc[(size_t)t0 * 768 + c8 + c] : 0.0f;
      v1[c] = (t1 < 90) ? vsrc[(size_t)t1 * 768 + c8 + c] : 0.0f;
    }
#pragma unroll
    for (int c = 0; c < 8; ++c)
      Vs2[(c8 + c) * 52 + p] = pack2_bf16(v0[c], v1[c]);
  }
  __syncthreads();

  // ---- S = Q @ K^T ----
  AFrag qf;
  {
    const __bf16* ap = Qs + (16 * wave + l) * 36 + hi * 8;
    qf.d[0] = *(const uint2*)(ap);
    qf.d[1] = *(const uint2*)(ap + 4);
    qf.d[2] = *(const uint2*)(ap + 16);
    qf.d[3] = *(const uint2*)(ap + 20);
  }
  floatx8 sc[6];
#pragma unroll
  for (int j = 0; j < 6; ++j) {
    BFragR kf;   // B[k][n] = K[16j+n][k] -> two 16B runs of K's row (k=8h.., 16+8h..)
    const __bf16* kp = Ks + (16 * j + l) * 36 + hi * 8;
    kf.d[0] = *(const uint2*)(kp);
    kf.d[1] = *(const uint2*)(kp + 4);
    kf.d[2] = *(const uint2*)(kp + 16);
    kf.d[3] = *(const uint2*)(kp + 20);
    floatx8 z;
#pragma unroll
    for (int e = 0; e < 8; ++e) z[e] = 0.0f;
    sc[j] = wmma_bf16(qf.v, kf.v, z);
  }

  // ---- mask + softmax in registers (row spans 16 lanes of a half-wave) ----
  const float scale = 0.17677669529663687f;   // 1/sqrt(32)
#pragma unroll
  for (int v = 0; v < 8; ++v) {
    const int rowTok = 16 * wave + 8 * hi + v;
    float m = -3.0e38f;
#pragma unroll
    for (int j = 0; j < 6; ++j) {
      const int colTok = 16 * j + l;
      float val = sc[j][v] * scale;
      if (colTok >= 90) val = -1.0e30f;               // padded keys
      else if (causal && colTok > rowTok) val = -1.0e9f;
      sc[j][v] = val;
      m = fmaxf(m, val);
    }
#pragma unroll
    for (int mm = 8; mm >= 1; mm >>= 1) m = fmaxf(m, __shfl_xor(m, mm, 32));
    float ssum = 0.0f;
#pragma unroll
    for (int j = 0; j < 6; ++j) {
      const float e = __expf(sc[j][v] - m);
      sc[j][v] = e;
      ssum += e;
    }
#pragma unroll
    for (int mm = 8; mm >= 1; mm >>= 1) ssum += __shfl_xor(ssum, mm, 32);
    const float inv = 1.0f / ssum;
#pragma unroll
    for (int j = 0; j < 6; ++j)
      Ps[rowTok * 104 + 16 * j + l] = to_bf16(sc[j][v] * inv);
  }
  __syncthreads();

  // ---- O = P @ V (K=96 -> 3 steps; 2 N-subtiles of 16) ----
  floatx8 o0, o1;
#pragma unroll
  for (int e = 0; e < 8; ++e) { o0[e] = 0.0f; o1[e] = 0.0f; }
#pragma unroll
  for (int t = 0; t < 3; ++t) {
    AFrag pf;
    const __bf16* pp = Ps + (16 * wave + l) * 104 + 32 * t + hi * 8;
    pf.d[0] = *(const uint2*)(pp);
    pf.d[1] = *(const uint2*)(pp + 4);
    pf.d[2] = *(const uint2*)(pp + 16);
    pf.d[3] = *(const uint2*)(pp + 20);
#pragma unroll
    for (int nsub = 0; nsub < 2; ++nsub) {
      BFragW vf;
      const unsigned int* vp = Vs2 + (16 * nsub + l) * 52 + 16 * t + hi * 4;
      vf.q[0] = *(const uint4*)(vp);
      vf.q[1] = *(const uint4*)(vp + 8);
      if (nsub == 0) o0 = wmma_bf16(pf.v, vf.v, o0);
      else           o1 = wmma_bf16(pf.v, vf.v, o1);
    }
  }
  // ---- store ----
#pragma unroll
  for (int nsub = 0; nsub < 2; ++nsub) {
#pragma unroll
    for (int v = 0; v < 8; ++v) {
      const int rowTok = 16 * wave + 8 * hi + v;
      if (rowTok < 90) {
        const float val = (nsub == 0) ? o0[v] : o1[v];
        out[((size_t)s * 90 + rowTok) * 256 + h * 32 + 16 * nsub + l] = val;
      }
    }
  }
}

// ---------------------------------------------------------------------------
// LayerNorm over D=256, one block (256 thr) per row; optional spike.
// ---------------------------------------------------------------------------
__global__ __launch_bounds__(256) void ln_kernel(
    const float* __restrict__ X, float* __restrict__ Y,
    const float* __restrict__ g, const float* __restrict__ b, int doSpike)
{
  const int row = blockIdx.x;
  const int t   = threadIdx.x;
  const float x = X[(size_t)row * 256 + t];
  __shared__ float red[8];

  float s = x;
#pragma unroll
  for (int m = 16; m >= 1; m >>= 1) s += __shfl_xor(s, m, 32);
  if ((t & 31) == 0) red[t >> 5] = s;
  __syncthreads();
  float mean = 0.0f;
#pragma unroll
  for (int i = 0; i < 8; ++i) mean += red[i];
  mean *= (1.0f / 256.0f);
  __syncthreads();

  const float d = x - mean;
  float q = d * d;
#pragma unroll
  for (int m = 16; m >= 1; m >>= 1) q += __shfl_xor(q, m, 32);
  if ((t & 31) == 0) red[t >> 5] = q;
  __syncthreads();
  float var = 0.0f;
#pragma unroll
  for (int i = 0; i < 8; ++i) var += red[i];
  var *= (1.0f / 256.0f);

  float y = d * rsqrtf(var + 1e-5f) * g[t] + b[t];
  if (doSpike) y = spike(y);
  Y[(size_t)row * 256 + t] = y;
}

// masks[t,d] = spike(mask_token[d] + pos[t,d])
__global__ void maskvec_kernel(const float* __restrict__ mtok,
                               const float* __restrict__ pos,
                               float* __restrict__ mv)
{
  const int t = blockIdx.x, d = threadIdx.x;
  mv[t * 256 + d] = spike(mtok[d] + pos[t * 256 + d]);
}

// Build predictor input chunk b: X[t*90+j, d] = (j==89) ? mask[t,d]
//   : student_full[b, j + (j>=t), d]
__global__ __launch_bounds__(256) void assemble_kernel(
    const float* __restrict__ student, const float* __restrict__ mv,
    float* __restrict__ X, int b)
{
  const int row = blockIdx.x;       // t*90 + j
  const int d   = threadIdx.x;
  const int t = row / 90, j = row % 90;
  float v;
  if (j == 89) v = mv[t * 256 + d];
  else {
    const int src = j + (j >= t ? 1 : 0);
    v = student[((size_t)b * 90 + src) * 256 + d];
  }
  X[(size_t)row * 256 + d] = v;
}

// pred_targets[b*90+t, d] = spike(X[t*90+89, d])
__global__ void gather_kernel(const float* __restrict__ X,
                              float* __restrict__ tgt, int b)
{
  const int t = blockIdx.x, d = threadIdx.x;
  tgt[((size_t)b * 90 + t) * 256 + d] = spike(X[((size_t)t * 90 + 89) * 256 + d]);
}

__global__ void copy_kernel(const float* __restrict__ src,
                            float* __restrict__ dst, int n)
{
  const int i = blockIdx.x * blockDim.x + threadIdx.x;
  if (i < n) dst[i] = src[i];
}

// means over last dim (240) for the three output blocks
__global__ __launch_bounds__(256) void means_kernel(const float* __restrict__ out,
                                                    float* __restrict__ means)
{
  const int row = blockIdx.x;     // 0..719
  const int t   = threadIdx.x;
  __shared__ float red[8];
#pragma unroll
  for (int k = 0; k < 3; ++k) {
    float v = (t < 240) ? out[(size_t)k * 172800 + (size_t)row * 240 + t] : 0.0f;
#pragma unroll
    for (int m = 16; m >= 1; m >>= 1) v += __shfl_xor(v, m, 32);
    if ((t & 31) == 0) red[t >> 5] = v;
    __syncthreads();
    if (t == 0) {
      float tot = 0.0f;
#pragma unroll
      for (int i = 0; i < 8; ++i) tot += red[i];
      means[k * 720 + row] = tot * (1.0f / 240.0f);
    }
    __syncthreads();
  }
}

// ---------------------------------------------------------------------------
// Host-side orchestration
// ---------------------------------------------------------------------------
static inline void launch_gemm(const float* A, const float* W, float* C,
                               int M, int N, int K,
                               const float* bias, const float* pos, int posR,
                               const float* resid, int doGelu, int doSpike,
                               hipStream_t stream)
{
  dim3 grid((M + 63) / 64, (N + 63) / 64);
  gemm_kernel<<<grid, 128, 0, stream>>>(A, W, C, M, N, K, bias, pos,
                                        posR <= 0 ? 1 : posR, resid,
                                        doGelu, doSpike);
}

static void run_layer(float* X, float* Xn, float* QKV, float* ATT, float* Hbuf,
                      int M, int numSeq,
                      const float* Wqkv, const float* Wo, const float* ln1,
                      const float* Wf1, const float* Wf2, const float* ln2,
                      int causal, hipStream_t stream)
{
  ln_kernel<<<M, 256, 0, stream>>>(X, Xn, ln1, ln1 + 256, 0);
  launch_gemm(Xn, Wqkv, QKV, M, 768, 256, nullptr, nullptr, 1, nullptr, 0, 0, stream);
  attention_kernel<<<dim3(numSeq, 8), 192, 0, stream>>>(QKV, ATT, causal);
  launch_gemm(ATT, Wo, X, M, 256, 256, nullptr, nullptr, 1, X, 0, 0, stream);
  ln_kernel<<<M, 256, 0, stream>>>(X, Xn, ln2, ln2 + 256, 0);
  launch_gemm(Xn, Wf1, Hbuf, M, 1024, 256, nullptr, nullptr, 1, nullptr, 1, 0, stream);
  launch_gemm(Hbuf, Wf2, X, M, 256, 1024, nullptr, nullptr, 1, X, 0, 0, stream);
}

extern "C" void kernel_launch(void* const* d_in, const int* in_sizes, int n_in,
                              void* d_out, int out_size, void* d_ws, size_t ws_size,
                              hipStream_t stream)
{
  (void)in_sizes; (void)n_in; (void)out_size;

  const float* x      = (const float*)d_in[0];   // (8,90,240)
  const float* W_emb  = (const float*)d_in[1];   // (240,256)
  const float* b_emb  = (const float*)d_in[2];   // (256)
  const float* pos    = (const float*)d_in[3];   // (1,90,256)
  const float* mtok   = (const float*)d_in[4];   // (1,1,256)
  const float* norm_g = (const float*)d_in[5];
  const float* norm_b = (const float*)d_in[6];
  const float* W_dec  = (const float*)d_in[7];   // (256,240)
  const float* b_dec  = (const float*)d_in[8];   // (240)
  const float* t_qkv = (const float*)d_in[9];
  const float* t_out = (const float*)d_in[10];
  const float* t_ln1 = (const float*)d_in[11];
  const float* t_ff1 = (const float*)d_in[12];
  const float* t_ff2 = (const float*)d_in[13];
  const float* t_ln2 = (const float*)d_in[14];
  const float* s_qkv = (const float*)d_in[15];
  const float* s_out = (const float*)d_in[16];
  const float* s_ln1 = (const float*)d_in[17];
  const float* s_ff1 = (const float*)d_in[18];
  const float* s_ff2 = (const float*)d_in[19];
  const float* s_ln2 = (const float*)d_in[20];
  const float* p_qkv = (const float*)d_in[21];
  const float* p_out = (const float*)d_in[22];
  const float* p_ln1 = (const float*)d_in[23];
  const float* p_ff1 = (const float*)d_in[24];
  const float* p_ff2 = (const float*)d_in[25];
  const float* p_ln2 = (const float*)d_in[26];

  // ---- workspace layout (floats) ----
  const size_t SZ_EMB  = 720 * 256;      // 184320
  const size_t SZ_MV   = 90 * 256;       // 23040
  const size_t SZ_X    = 8100 * 256;     // chunk activations
  const size_t SZ_QKV  = 8100 * 768;
  const size_t SZ_H    = 8100 * 1024;
  const size_t NEED = 4 * SZ_EMB + SZ_MV + 3 * SZ_X + SZ_QKV + SZ_H;
  if (ws_size < NEED * sizeof(float)) return;

  float* ws           = (float*)d_ws;
  float* x_emb        = ws;
  float* teacher_full = x_emb + SZ_EMB;
  float* student_full = teacher_full + SZ_EMB;
  float* mv           = student_full + SZ_EMB;
  float* pred_tgt     = mv + SZ_MV;
  float* X            = pred_tgt + SZ_EMB;
  float* Xn           = X + SZ_X;
  float* QKV          = Xn + SZ_X;
  float* ATT          = QKV + SZ_QKV;
  float* Hbuf         = ATT + SZ_X;

  float* out = (float*)d_out;

  // 1) x_emb = spike(x @ W_embed + b_embed + pos)
  launch_gemm(x, W_emb, x_emb, 720, 256, 240, b_emb, pos, 90, nullptr, 0, 1, stream);

  // 2) teacher encoder (8 layers, non-causal) -> spike(LN(.))
  hipMemcpyAsync(X, x_emb, SZ_EMB * sizeof(float), hipMemcpyDeviceToDevice, stream);
  for (int i = 0; i < 8; ++i)
    run_layer(X, Xn, QKV, ATT, Hbuf, 720, 8,
              t_qkv + (size_t)i * 256 * 768, t_out + (size_t)i * 256 * 256,
              t_ln1 + (size_t)i * 512,
              t_ff1 + (size_t)i * 256 * 1024, t_ff2 + (size_t)i * 1024 * 256,
              t_ln2 + (size_t)i * 512, 0, stream);
  ln_kernel<<<720, 256, 0, stream>>>(X, teacher_full, norm_g, norm_b, 1);

  // 3) student encoder
  hipMemcpyAsync(X, x_emb, SZ_EMB * sizeof(float), hipMemcpyDeviceToDevice, stream);
  for (int i = 0; i < 8; ++i)
    run_layer(X, Xn, QKV, ATT, Hbuf, 720, 8,
              s_qkv + (size_t)i * 256 * 768, s_out + (size_t)i * 256 * 256,
              s_ln1 + (size_t)i * 512,
              s_ff1 + (size_t)i * 256 * 1024, s_ff2 + (size_t)i * 1024 * 256,
              s_ln2 + (size_t)i * 512, 0, stream);
  ln_kernel<<<720, 256, 0, stream>>>(X, student_full, norm_g, norm_b, 1);

  // 4) mask vectors
  maskvec_kernel<<<90, 256, 0, stream>>>(mtok, pos, mv);

  // 5) predictor: 8 chunks of 90 sequences x 90 tokens (causal, 6 layers)
  for (int b = 0; b < 8; ++b) {
    assemble_kernel<<<8100, 256, 0, stream>>>(student_full, mv, X, b);
    for (int i = 0; i < 6; ++i)
      run_layer(X, Xn, QKV, ATT, Hbuf, 8100, 90,
                p_qkv + (size_t)i * 256 * 768, p_out + (size_t)i * 256 * 256,
                p_ln1 + (size_t)i * 512,
                p_ff1 + (size_t)i * 256 * 1024, p_ff2 + (size_t)i * 1024 * 256,
                p_ln2 + (size_t)i * 512, 1, stream);
    gather_kernel<<<90, 256, 0, stream>>>(X, pred_tgt, b);
  }

  // 6) decode: pred_seq / teacher_seq = spike(. @ W_dec + b_dec)
  launch_gemm(pred_tgt, W_dec, out, 720, 240, 256, b_dec, nullptr, 1, nullptr, 0, 1, stream);
  launch_gemm(teacher_full, W_dec, out + 172800, 720, 240, 256, b_dec, nullptr, 1, nullptr, 0, 1, stream);

  // 7) target_seq = x; then the three means
  copy_kernel<<<(172800 + 255) / 256, 256, 0, stream>>>(x, out + 345600, 172800);
  means_kernel<<<720, 256, 0, stream>>>(out, out + 518400);
}
